// Block_29360396436026
// MI455X (gfx1250) — compile-verified, matrix operand order checked by reading, not run on hardware
//
#include <hip/hip_runtime.h>

typedef __attribute__((ext_vector_type(16))) _Float16 v16h;
typedef __attribute__((ext_vector_type(8)))  float    v8f;

#define NODE_DIM 64
#define EDGE_DIM 32
#define HIDDEN   64
#define WPB      8        // waves per block (256 threads, wave32)
#define HPAD     72       // padded row stride (floats) for h staging

// Build a 16x32 f16 A-fragment for this lane from 16 consecutive floats
// at p0 (K = base..base+7) and p0+16 (K = base+16..base+23).
// v16h element j -> within-fragment K = j (j<8) / j+8 (j>=8), lane-half offset
// is already folded into p0 by the caller (hi*8).
__device__ __forceinline__ v16h build_afrag(const float* p0) {
    float4 u0 = ((const float4*)p0)[0];
    float4 u1 = ((const float4*)p0)[1];
    float4 w0 = ((const float4*)(p0 + 16))[0];
    float4 w1 = ((const float4*)(p0 + 16))[1];
    float t[16] = {u0.x,u0.y,u0.z,u0.w, u1.x,u1.y,u1.z,u1.w,
                   w0.x,w0.y,w0.z,w0.w, w1.x,w1.y,w1.z,w1.w};
    v16h a;
#pragma unroll
    for (int j = 0; j < 16; ++j) a[j] = (_Float16)t[j];
    return a;
}

__global__ __launch_bounds__(256) void gnn_edge_mlp_scatter(
    const float* __restrict__ x,        // [N,64]
    const int*   __restrict__ dst,      // edge_index row 1, [E]
    const float* __restrict__ ea,       // [E,32]
    const float* __restrict__ W1,       // [96,64]
    const float* __restrict__ b1,       // [64]
    const float* __restrict__ W2,       // [64,64]
    const float* __restrict__ b2,       // [64]
    float* __restrict__ outsum,         // [N,64] accumulator (pre-zeroed)
    float* __restrict__ cnt,            // [N] float counts (pre-zeroed)
    int E, int numTiles)
{
    // f16 weights pre-swizzled into per-lane WMMA B-fragment order:
    // fragment f = ks*4+nt, then [lane][16 halves], 32B per lane.
    __shared__ __align__(32) _Float16 wB1[12 * 32 * 16]; // 96x64
    __shared__ __align__(32) _Float16 wB2[ 8 * 32 * 16]; // 64x64
    __shared__ __align__(16) float    hbuf[WPB][16 * HPAD];
    __shared__ int dstbuf[WPB][16];

    const int tid = threadIdx.x;

    // ---- stage W1 ----
    for (int idx = tid; idx < 12 * 32 * 16; idx += blockDim.x) {
        int j = idx & 15, lane = (idx >> 4) & 31, f = idx >> 9;
        int ks = f >> 2, nt = f & 3, hi = lane >> 4;
        int k = ks * 32 + (j < 8 ? j : j + 8) + hi * 8;
        int n = nt * 16 + (lane & 15);
        wB1[idx] = (_Float16)W1[k * 64 + n];
    }
    // ---- stage W2 ----
    for (int idx = tid; idx < 8 * 32 * 16; idx += blockDim.x) {
        int j = idx & 15, lane = (idx >> 4) & 31, f = idx >> 9;
        int ks = f >> 2, nt = f & 3, hi = lane >> 4;
        int k = ks * 32 + (j < 8 ? j : j + 8) + hi * 8;
        int n = nt * 16 + (lane & 15);
        wB2[idx] = (_Float16)W2[k * 64 + n];
    }
    __syncthreads();

    const int lane = tid & 31;
    const int wave = tid >> 5;
    const int hi   = lane >> 4;      // which half-wave
    const int r    = lane & 15;      // row (edge-in-tile) / column-in-tile id
    float* hb = &hbuf[wave][0];
    int*   db = &dstbuf[wave][0];

    const int gwave   = blockIdx.x * WPB + wave;
    const int wstride = gridDim.x * WPB;

    for (int t = gwave; t < numTiles; t += wstride) {
        const int e  = t * 16 + r;
        const int ec = e < E ? e : (E - 1);
        const int dr = dst[ec];
        if (lane < 16) db[r] = dr;

        // ================= GEMM1: [16x96] @ [96x64] =================
        v8f acc[4];
#pragma unroll
        for (int nt = 0; nt < 4; ++nt)
#pragma unroll
            for (int i = 0; i < 8; ++i) acc[nt][i] = 0.0f;

#pragma unroll
        for (int ks = 0; ks < 3; ++ks) {
            const float* p0 = (ks < 2)
                ? (x  + (size_t)dr * NODE_DIM + ks * 32 + hi * 8)
                : (ea + (size_t)ec * EDGE_DIM + hi * 8);
            v16h a = build_afrag(p0);
#pragma unroll
            for (int nt = 0; nt < 4; ++nt) {
                v16h b = *(const v16h*)&wB1[((ks * 4 + nt) * 32 + lane) * 16];
                acc[nt] = __builtin_amdgcn_wmma_f32_16x16x32_f16(
                    false, a, false, b, (short)0, acc[nt], false, false);
            }
        }

        // bias + ReLU, stage h (row-major [16][HPAD]) for the transpose
#pragma unroll
        for (int nt = 0; nt < 4; ++nt) {
            const int   n    = nt * 16 + r;
            const float bias = b1[n];
#pragma unroll
            for (int i = 0; i < 8; ++i) {
                float v = acc[nt][i] + bias;
                v = v > 0.0f ? v : 0.0f;
                hb[(i + hi * 8) * HPAD + n] = v;   // row = M, col = feature
            }
        }
        asm volatile("s_wait_dscnt 0" ::: "memory");

        // ================= GEMM2: [16x64] @ [64x64] =================
        v8f acc2[4];
#pragma unroll
        for (int nt = 0; nt < 4; ++nt)
#pragma unroll
            for (int i = 0; i < 8; ++i) acc2[nt][i] = 0.0f;

#pragma unroll
        for (int ks = 0; ks < 2; ++ks) {
            const float* p0 = hb + r * HPAD + ks * 32 + hi * 8;
            v16h a = build_afrag(p0);
#pragma unroll
            for (int nt = 0; nt < 4; ++nt) {
                v16h b = *(const v16h*)&wB2[((ks * 4 + nt) * 32 + lane) * 16];
                acc2[nt] = __builtin_amdgcn_wmma_f32_16x16x32_f16(
                    false, a, false, b, (short)0, acc2[nt], false, false);
            }
        }

        // bias + ReLU + atomic scatter (segment sum into L2-resident acc)
#pragma unroll
        for (int nt = 0; nt < 4; ++nt) {
            const int   n    = nt * 16 + r;
            const float bias = b2[n];
#pragma unroll
            for (int i = 0; i < 8; ++i) {
                float v = acc2[nt][i] + bias;
                v = v > 0.0f ? v : 0.0f;
                const int m  = i + hi * 8;      // edge row in tile
                const int em = t * 16 + m;
                if (em < E) {
                    const int dm = db[m];
                    __hip_atomic_fetch_add(&outsum[(size_t)dm * NODE_DIM + n], v,
                                           __ATOMIC_RELAXED, __HIP_MEMORY_SCOPE_AGENT);
                }
            }
        }
        if (lane < 16 && e < E) {
            __hip_atomic_fetch_add(&cnt[dr], 1.0f,
                                   __ATOMIC_RELAXED, __HIP_MEMORY_SCOPE_AGENT);
        }
    }
}

__global__ void zero_kernel(float* __restrict__ out, int n1,
                            float* __restrict__ c, int n2) {
    int i = blockIdx.x * blockDim.x + threadIdx.x;
    int s = gridDim.x * blockDim.x;
    for (int k = i; k < n1; k += s) out[k] = 0.0f;
    for (int k = i; k < n2; k += s) c[k] = 0.0f;
}

__global__ void finalize_kernel(float* __restrict__ out,
                                const float* __restrict__ cnt,
                                const float* __restrict__ x, int total) {
    int i = blockIdx.x * blockDim.x + threadIdx.x;
    if (i < total) {
        float c = cnt[i >> 6];
        out[i] = out[i] / fmaxf(c, 1.0f) + x[i];
    }
}

extern "C" void kernel_launch(void* const* d_in, const int* in_sizes, int n_in,
                              void* d_out, int out_size, void* d_ws, size_t ws_size,
                              hipStream_t stream) {
    const float* x   = (const float*)d_in[0];
    const int*   ei  = (const int*)  d_in[1];   // [2,E] int32
    const float* ea  = (const float*)d_in[2];
    const float* W1  = (const float*)d_in[3];
    const float* b1  = (const float*)d_in[4];
    const float* W2  = (const float*)d_in[5];
    const float* b2  = (const float*)d_in[6];

    const int N = in_sizes[0] / NODE_DIM;
    const int E = in_sizes[1] / 2;
    const int* dst = ei + E;                    // row 1 of edge_index

    float* outsum = (float*)d_out;              // [N,64] accumulator == output
    float* cnt    = (float*)d_ws;               // [N] counts

    const int total = N * NODE_DIM;

    {
        int blocks = (total + 255) / 256;
        if (blocks > 4096) blocks = 4096;
        zero_kernel<<<blocks, 256, 0, stream>>>(outsum, total, cnt, N);
    }

    const int numTiles = (E + 15) / 16;
    {
        int blocks = (numTiles + WPB - 1) / WPB;
        if (blocks > 4096) blocks = 4096;
        gnn_edge_mlp_scatter<<<blocks, 256, 0, stream>>>(
            x, dst, ea, W1, b1, W2, b2, outsum, cnt, E, numTiles);
    }

    {
        int blocks = (total + 255) / 256;
        finalize_kernel<<<blocks, 256, 0, stream>>>(outsum, cnt, x, total);
    }
}
